// MF_IPM_Layer_69492570849369
// MI455X (gfx1250) — matrix operand order repaired and checked by reading, not compiled, not run on hardware
//
#include <hip/hip_runtime.h>
#include <hip/hip_bf16.h>
#include <math.h>

// MI455X / gfx1250 implementation of the multi-scale bidirectional Mamba block.
// GEMMs: V_WMMA_F32_16X16X32_F16 core, B-tiles fetched by the Tensor Data Mover
// (tensor_load_to_lds + s_wait_tensorcnt), A-tiles staged as packed f16 in LDS.

typedef __attribute__((ext_vector_type(16))) _Float16 v16h;
typedef __attribute__((ext_vector_type(8)))  _Float16 v8h;
typedef __attribute__((ext_vector_type(8)))  float    v8f;
typedef unsigned int u32x4 __attribute__((ext_vector_type(4)));
typedef int          i32x4 __attribute__((ext_vector_type(4)));
typedef int          i32x8 __attribute__((ext_vector_type(8)));

#define DEVI __device__ __forceinline__

DEVI float silu_f(float x)      { return x / (1.0f + __expf(-x)); }
DEVI float softplus_f(float x)  { return (x > 20.0f) ? x : log1pf(__expf(x)); }

// ---------------------------------------------------------------------------
// Tensor Data Mover: async 2D f32 tile load (global -> LDS), ISA 8.3-8.5.
// tile: tile_d1 rows x tile_d0 elements, row stride = stride_elems.
// Rows/cols past (dim0_rem, dim1_rem) are zero-filled by hardware OOB.
// 6-arg builtin form (clang-23 / therock headers): g0, g1, g2, g3, extra, cpol.
// ---------------------------------------------------------------------------
DEVI void tdm_load_2d_f32(unsigned lds_byte_addr, const void* gaddr,
                          int dim0_rem, int dim1_rem, int stride_elems,
                          int tile_d0, int tile_d1) {
    unsigned long long ga = (unsigned long long)(uintptr_t)gaddr;
    u32x4 g0;
    g0[0] = 1u;                                              // count=1 (user D#)
    g0[1] = lds_byte_addr;                                   // lds_addr
    g0[2] = (unsigned)(ga & 0xFFFFFFFFu);                    // global_addr lo
    g0[3] = (unsigned)((ga >> 32) & 0x01FFFFFFu) | (2u << 30); // addr hi | type=2
    i32x8 g1;
    g1[0] = (2 << 16);                                       // data_size = 4B
    g1[1] = (dim0_rem & 0xFFFF) << 16;                       // tensor_dim0 lo16
    g1[2] = ((dim0_rem >> 16) & 0xFFFF) | ((dim1_rem & 0xFFFF) << 16); // d0 hi | d1 lo
    g1[3] = ((dim1_rem >> 16) & 0xFFFF) | (tile_d0 << 16);   // d1 hi | tile_dim0
    g1[4] = (tile_d1 & 0xFFFF);                              // tile_dim1, tile_dim2=0
    g1[5] = stride_elems;                                    // tensor_dim0_stride lo32
    g1[6] = 0;
    g1[7] = 0;
    i32x4 z4 = {0, 0, 0, 0};
    i32x8 z8 = {0, 0, 0, 0, 0, 0, 0, 0};
    __builtin_amdgcn_tensor_load_to_lds(g0, g1, z4, z4, z8, 0);
}

// ---------------------------------------------------------------------------
// Block reduction (wave32-aware)
// ---------------------------------------------------------------------------
DEVI float block_reduce_sum(float v, float* red) {
    int lane = threadIdx.x & 31, wv = threadIdx.x >> 5;
    #pragma unroll
    for (int o = 16; o > 0; o >>= 1) v += __shfl_down(v, o, 32);
    if (lane == 0) red[wv] = v;
    __syncthreads();
    int nw = blockDim.x >> 5;
    v = (threadIdx.x < (unsigned)nw) ? red[threadIdx.x] : 0.0f;
    if (wv == 0) {
        #pragma unroll
        for (int o = 16; o > 0; o >>= 1) v += __shfl_down(v, o, 32);
    }
    if (threadIdx.x == 0) red[0] = v;
    __syncthreads();
    float r = red[0];
    __syncthreads();
    return r;
}

// ---------------------------------------------------------------------------
// LayerNorm over last dim C. One block per row.
// ---------------------------------------------------------------------------
__global__ void ln_kernel(const float* __restrict__ x, const float* __restrict__ w,
                          const float* __restrict__ b, float* __restrict__ out, int C) {
    __shared__ float red[32];
    size_t row = blockIdx.x;
    const float* xr = x + row * (size_t)C;
    float* orow = out + row * (size_t)C;
    float s = 0.0f;
    for (int c = threadIdx.x; c < C; c += blockDim.x) s += xr[c];
    float mu = block_reduce_sum(s, red) / (float)C;
    float s2 = 0.0f;
    for (int c = threadIdx.x; c < C; c += blockDim.x) {
        float dv = xr[c] - mu; s2 += dv * dv;
    }
    float var = block_reduce_sum(s2, red) / (float)C;
    float inv = rsqrtf(var + 1e-5f);
    for (int c = threadIdx.x; c < C; c += blockDim.x)
        orow[c] = (xr[c] - mu) * inv * w[c] + b[c];
}

// ---------------------------------------------------------------------------
// Time-average pool by df:  out (B,L/df,C) = mean over df rows of xn (B,L,C)
// ---------------------------------------------------------------------------
__global__ void pool_kernel(const float* __restrict__ xn, float* __restrict__ outp,
                            int B, int L, int C, int df) {
    int Lo = L / df;
    size_t total = (size_t)B * Lo * C;
    size_t idx = (size_t)blockIdx.x * blockDim.x + threadIdx.x;
    if (idx >= total) return;
    int c = (int)(idx % C);
    size_t bt = idx / C;
    int t = (int)(bt % Lo);
    int b = (int)(bt / Lo);
    float s = 0.0f;
    for (int j = 0; j < df; ++j)
        s += xn[((size_t)b * L + (size_t)t * df + j) * C + c];
    outp[idx] = s / (float)df;
}

// ---------------------------------------------------------------------------
// GEMM: O[m,n] = act( sum_k A[m,k] * W[n,k] ), f32 IO, f16 WMMA core.
// Block = 256 threads = 8 waves. Block tile 32(M) x 128(N), k-step 32.
// Waves: wm = wv&1 (2 M-tiles), wn = wv>>1 (4 N-groups), 2 N-subtiles each
// -> two v_wmma_f32_16x16x32_f16 per wave per k-step.
// A tile: vector-staged packed f16 in LDS (contiguous ds_load_b128 gathers).
// B tile: TDM tensor_load_to_lds (f32, hardware OOB zero-fill for ragged N).
// Requires: M % 32 == 0, K % 32 == 0. N arbitrary.
// ---------------------------------------------------------------------------
__global__ __launch_bounds__(256)
void gemm_wmma_kernel(const float* __restrict__ A, const float* __restrict__ W,
                      float* __restrict__ O, int M, int N, int K, int act) {
    __shared__ _Float16 As[32 * 32];   // 2 KB  : As[row][k] packed f16
    __shared__ float    BsF[128 * 32]; // 16 KB : BsF[n][k] f32 (TDM destination)
    const int tid  = threadIdx.x;
    const int wv   = tid >> 5;
    const int lane = tid & 31;
    const int hi = lane >> 4, lm = lane & 15;
    const int wm = wv & 1, wn = wv >> 1;
    const int m0 = blockIdx.x * 32;
    const int n0 = blockIdx.y * 128;
    const int arow = wm * 16 + lm;
    const int nl0 = (wn * 2 + 0) * 16 + lm;
    const int nl1 = (wn * 2 + 1) * 16 + lm;
    const int wv_s = __builtin_amdgcn_readfirstlane(wv);
    const unsigned bs_lds = (unsigned)(uintptr_t)(void*)&BsF[0];

    v8f acc0 = {}, acc1 = {};
    for (int k0 = 0; k0 < K; k0 += 32) {
        // ---- B tile via Tensor Data Mover (wave 0 only; uniform branch) ----
        if (wv_s == 0) {
            tdm_load_2d_f32(bs_lds, W + (size_t)n0 * K + k0,
                            K - k0, N - n0, K, 32, 128);
        }
        // ---- A tile: 32x32 f32 -> packed f16, 4 elems/thread ----
        {
            int r = tid >> 3;            // 0..31
            int c = (tid & 7) * 4;       // 0..28
            if (k0 + 32 < K)
                __builtin_prefetch(&A[(size_t)(m0 + r) * K + k0 + 32 + c], 0, 1);
            float4 f = *(const float4*)&A[(size_t)(m0 + r) * K + k0 + c];
            union { _Float16 h[4]; uint2 u; } p;
            p.h[0] = (_Float16)f.x; p.h[1] = (_Float16)f.y;
            p.h[2] = (_Float16)f.z; p.h[3] = (_Float16)f.w;
            *(uint2*)&As[r * 32 + c] = p.u;
        }
        if (wv_s == 0) __builtin_amdgcn_s_wait_tensorcnt(0);
        __syncthreads();

        // ---- Gather A operand: 16 consecutive halves (k = 16*hi + s) ----
        union { v16h v; v8h h8[2]; } au;
        au.h8[0] = *(const v8h*)&As[arow * 32 + 16 * hi];
        au.h8[1] = *(const v8h*)&As[arow * 32 + 16 * hi + 8];

        // ---- Gather B operands: 16 consecutive f32 per subtile, convert ----
        {
            const float4* bp = (const float4*)&BsF[nl0 * 32 + 16 * hi];
            float4 q0 = bp[0], q1 = bp[1], q2 = bp[2], q3 = bp[3];
            v16h bv;
            bv[0]=(_Float16)q0.x; bv[1]=(_Float16)q0.y; bv[2]=(_Float16)q0.z; bv[3]=(_Float16)q0.w;
            bv[4]=(_Float16)q1.x; bv[5]=(_Float16)q1.y; bv[6]=(_Float16)q1.z; bv[7]=(_Float16)q1.w;
            bv[8]=(_Float16)q2.x; bv[9]=(_Float16)q2.y; bv[10]=(_Float16)q2.z; bv[11]=(_Float16)q2.w;
            bv[12]=(_Float16)q3.x; bv[13]=(_Float16)q3.y; bv[14]=(_Float16)q3.z; bv[15]=(_Float16)q3.w;
            acc0 = __builtin_amdgcn_wmma_f32_16x16x32_f16(
                       false, au.v, false, bv, (short)0, acc0, false, false);
        }
        {
            const float4* bp = (const float4*)&BsF[nl1 * 32 + 16 * hi];
            float4 q0 = bp[0], q1 = bp[1], q2 = bp[2], q3 = bp[3];
            v16h bv;
            bv[0]=(_Float16)q0.x; bv[1]=(_Float16)q0.y; bv[2]=(_Float16)q0.z; bv[3]=(_Float16)q0.w;
            bv[4]=(_Float16)q1.x; bv[5]=(_Float16)q1.y; bv[6]=(_Float16)q1.z; bv[7]=(_Float16)q1.w;
            bv[8]=(_Float16)q2.x; bv[9]=(_Float16)q2.y; bv[10]=(_Float16)q2.z; bv[11]=(_Float16)q2.w;
            bv[12]=(_Float16)q3.x; bv[13]=(_Float16)q3.y; bv[14]=(_Float16)q3.z; bv[15]=(_Float16)q3.w;
            acc1 = __builtin_amdgcn_wmma_f32_16x16x32_f16(
                       false, au.v, false, bv, (short)0, acc1, false, false);
        }
        __syncthreads();
    }
    // D layout: lanes 0-15: M=r, lanes 16-31: M=r+8 ; N = lane&15
    #pragma unroll
    for (int r = 0; r < 8; ++r) {
        int m = m0 + wm * 16 + r + 8 * hi;
        if (m < M) {
            int na = n0 + nl0, nb = n0 + nl1;
            float va = acc0[r], vb = acc1[r];
            if (act == 1) { va = fmaxf(va, 0.0f); vb = fmaxf(vb, 0.0f); }
            if (na < N) O[(size_t)m * N + na] = va;
            if (nb < N) O[(size_t)m * N + nb] = vb;
        }
    }
}

// ---------------------------------------------------------------------------
// Copy with optional time flip: out(B,Lx,D) = in(B, flip? Lx-1-t : t, D)
// ---------------------------------------------------------------------------
__global__ void flipcopy_kernel(const float* __restrict__ in, float* __restrict__ out,
                                int B, int Lx, int D, int flip) {
    size_t total = (size_t)B * Lx * D;
    size_t idx = (size_t)blockIdx.x * blockDim.x + threadIdx.x;
    if (idx >= total) return;
    int d = (int)(idx % D);
    size_t bt = idx / D;
    int t = (int)(bt % Lx);
    int b = (int)(bt / Lx);
    int ts = flip ? (Lx - 1 - t) : t;
    out[idx] = in[((size_t)b * Lx + ts) * D + d];
}

// ---------------------------------------------------------------------------
// Depthwise causal conv (d_conv=4) + bias + SiLU on xc = xz[..., 0:256]
// ---------------------------------------------------------------------------
__global__ void conv_silu_kernel(const float* __restrict__ xz, const float* __restrict__ cw,
                                 const float* __restrict__ cb, float* __restrict__ out,
                                 int B, int Lx) {
    size_t total = (size_t)B * Lx * 256;
    size_t idx = (size_t)blockIdx.x * blockDim.x + threadIdx.x;
    if (idx >= total) return;
    int d = (int)(idx % 256);
    size_t bt = idx / 256;
    int t = (int)(bt % Lx);
    int b = (int)(bt / Lx);
    float s = cb[d];
    #pragma unroll
    for (int j = 0; j < 4; ++j) {
        int ts = t - 3 + j;
        if (ts >= 0) s += cw[d * 4 + j] * xz[((size_t)b * Lx + ts) * 512 + d];
    }
    out[idx] = silu_f(s);
}

// ---------------------------------------------------------------------------
// dt = softplus(xdbl[:, 0:8] @ dt_w.T + dt_b) ; K=8, pure VALU
// ---------------------------------------------------------------------------
__global__ void dt_kernel(const float* __restrict__ xdbl, const float* __restrict__ dtw,
                          const float* __restrict__ dtb, float* __restrict__ out,
                          size_t Mrows) {
    size_t total = Mrows * 256;
    size_t idx = (size_t)blockIdx.x * blockDim.x + threadIdx.x;
    if (idx >= total) return;
    int d = (int)(idx % 256);
    size_t r = idx / 256;
    const float* xr = xdbl + r * 40;
    float s = dtb[d];
    #pragma unroll
    for (int j = 0; j < 8; ++j) s += xr[j] * dtw[d * 8 + j];
    out[idx] = softplus_f(s);
}

// ---------------------------------------------------------------------------
// Selective scan. Block = one batch element (256 threads = 256 channels).
// Each lane keeps 16 hidden states + A row in registers; Bm/Cm staged in LDS.
// ---------------------------------------------------------------------------
__global__ void scan_kernel(const float* __restrict__ dt, const float* __restrict__ u,
                            const float* __restrict__ xdbl, const float* __restrict__ A_log,
                            const float* __restrict__ Dp, float* __restrict__ y, int Lx) {
    int b = blockIdx.x;
    int d = threadIdx.x;
    float A[16], h[16];
    #pragma unroll
    for (int n = 0; n < 16; ++n) { A[n] = -__expf(A_log[d * 16 + n]); h[n] = 0.0f; }
    float Dv = Dp[d];
    __shared__ float BmS[32 * 16];
    __shared__ float CmS[32 * 16];
    for (int t0 = 0; t0 < Lx; t0 += 32) {
        for (int e = threadIdx.x; e < 512; e += 256) {
            int tt = e >> 4, nn = e & 15;
            const float* row = xdbl + ((size_t)b * Lx + t0 + tt) * 40;
            BmS[e] = row[8 + nn];
            CmS[e] = row[24 + nn];
        }
        __syncthreads();
        for (int tt = 0; tt < 32; ++tt) {
            size_t gi = ((size_t)b * Lx + t0 + tt) * 256 + d;
            float dtv = dt[gi], uv = u[gi];
            float du = dtv * uv;
            float yv = 0.0f;
            #pragma unroll
            for (int n = 0; n < 16; ++n) {
                float dA = __expf(dtv * A[n]);
                h[n] = dA * h[n] + du * BmS[tt * 16 + n];
                yv += h[n] * CmS[tt * 16 + n];
            }
            y[gi] = yv + uv * Dv;
        }
        __syncthreads();
    }
}

// ---------------------------------------------------------------------------
// g = y * silu(z), z = xz[..., 256:512]
// ---------------------------------------------------------------------------
__global__ void gate_kernel(const float* __restrict__ y, const float* __restrict__ xz,
                            float* __restrict__ g, size_t Mrows) {
    size_t total = Mrows * 256;
    size_t idx = (size_t)blockIdx.x * blockDim.x + threadIdx.x;
    if (idx >= total) return;
    int d = (int)(idx % 256);
    size_t r = idx / 256;
    g[idx] = y[idx] * silu_f(xz[r * 512 + 256 + d]);
}

// ---------------------------------------------------------------------------
// yacc = ydir (dir 0)  |  yacc += time-flipped ydir (dir 1)
// ---------------------------------------------------------------------------
__global__ void accum_kernel(const float* __restrict__ ydir, float* __restrict__ yacc,
                             int B, int Lx, int flip) {
    size_t total = (size_t)B * Lx * 128;
    size_t idx = (size_t)blockIdx.x * blockDim.x + threadIdx.x;
    if (idx >= total) return;
    int d = (int)(idx % 128);
    size_t bt = idx / 128;
    int t = (int)(bt % Lx);
    int b = (int)(bt / Lx);
    int ts = flip ? (Lx - 1 - t) : t;
    float v = ydir[((size_t)b * Lx + ts) * 128 + d];
    if (flip) yacc[idx] += v; else yacc[idx] = v;
}

// ---------------------------------------------------------------------------
// out_pre[b,t, scale*128+d] = yacc[b,t/df,d]*out_attn[b,scale,d] + xi[b,t/df,d]*skip
// ---------------------------------------------------------------------------
__global__ void combine_kernel(const float* __restrict__ yacc, const float* __restrict__ xi,
                               const float* __restrict__ out_attn, const float* __restrict__ skip,
                               float* __restrict__ out_pre, int B, int L, int Lx,
                               int df, int scale) {
    size_t total = (size_t)B * L * 128;
    size_t idx = (size_t)blockIdx.x * blockDim.x + threadIdx.x;
    if (idx >= total) return;
    int d = (int)(idx % 128);
    size_t bt = idx / 128;
    int t = (int)(bt % L);
    int b = (int)(bt / L);
    int ti = t / df;
    float a  = out_attn[((size_t)b * 4 + scale) * 128 + d];
    size_t src = ((size_t)b * Lx + ti) * 128 + d;
    float v = yacc[src] * a + xi[src] * skip[0];
    out_pre[((size_t)b * L + t) * 512 + scale * 128 + d] = v;
}

// ---------------------------------------------------------------------------
// Channel attention: avg/max over time, shared MLP, sum. grid = B, block = 128.
// ---------------------------------------------------------------------------
__global__ void chattn_kernel(const float* __restrict__ xh, const float* __restrict__ fc1,
                              const float* __restrict__ fc2, float* __restrict__ Mout,
                              int Lx, int scale) {
    int b = blockIdx.x, d = threadIdx.x;
    __shared__ float avg[128], mx[128], h1[128];
    float s = 0.0f, m = -1e30f;
    for (int t = 0; t < Lx; ++t) {
        float v = xh[((size_t)b * Lx + t) * 128 + d];
        s += v; m = fmaxf(m, v);
    }
    avg[d] = s / (float)Lx;
    mx[d]  = m;
    __syncthreads();
    float acc = 0.0f;
    for (int k = 0; k < 128; ++k) acc += fc1[d * 128 + k] * avg[k];
    h1[d] = fmaxf(acc, 0.0f);
    __syncthreads();
    float oa = 0.0f;
    for (int k = 0; k < 128; ++k) oa += fc2[d * 128 + k] * h1[k];
    __syncthreads();
    acc = 0.0f;
    for (int k = 0; k < 128; ++k) acc += fc1[d * 128 + k] * mx[k];
    h1[d] = fmaxf(acc, 0.0f);
    __syncthreads();
    float om = 0.0f;
    for (int k = 0; k < 128; ++k) om += fc2[d * 128 + k] * h1[k];
    Mout[((size_t)b * 4 + scale) * 128 + d] = oa + om;
}

// ---------------------------------------------------------------------------
// Scale attention: out_attn = softmax(M M^T) M ; grid = B, block = 128.
// ---------------------------------------------------------------------------
__global__ void sattn_kernel(const float* __restrict__ Mm, float* __restrict__ out_attn) {
    int b = blockIdx.x, t = threadIdx.x;
    __shared__ float P[4][4];
    if (t < 16) {
        int i = t >> 2, j = t & 3;
        float s = 0.0f;
        for (int k = 0; k < 128; ++k)
            s += Mm[((size_t)b * 4 + i) * 128 + k] * Mm[((size_t)b * 4 + j) * 128 + k];
        P[i][j] = s;
    }
    __syncthreads();
    if (t < 4) {
        int i = t;
        float mxv = fmaxf(fmaxf(P[i][0], P[i][1]), fmaxf(P[i][2], P[i][3]));
        float e0 = __expf(P[i][0] - mxv), e1 = __expf(P[i][1] - mxv);
        float e2 = __expf(P[i][2] - mxv), e3 = __expf(P[i][3] - mxv);
        float sm = e0 + e1 + e2 + e3;
        P[i][0] = e0 / sm; P[i][1] = e1 / sm; P[i][2] = e2 / sm; P[i][3] = e3 / sm;
    }
    __syncthreads();
    #pragma unroll
    for (int i = 0; i < 4; ++i) {
        float s = 0.0f;
        #pragma unroll
        for (int j = 0; j < 4; ++j)
            s += P[i][j] * Mm[((size_t)b * 4 + j) * 128 + t];
        out_attn[((size_t)b * 4 + i) * 128 + t] = s;
    }
}

// ---------------------------------------------------------------------------
// Host launcher
// ---------------------------------------------------------------------------
extern "C" void kernel_launch(void* const* d_in, const int* in_sizes, int n_in,
                              void* d_out, int out_size, void* d_ws, size_t ws_size,
                              hipStream_t stream) {
    (void)in_sizes; (void)n_in; (void)out_size; (void)ws_size;
    const int B = 4, L = 4096, C = 512, DM = 128, DI = 256;
    const int Li[4] = {4096, 2048, 1024, 1024};
    const int df[4] = {1, 2, 4, 4};

    const float* x       = (const float*)d_in[0];
    const float* norm_w  = (const float*)d_in[1];
    const float* norm_b  = (const float*)d_in[2];
    const float* skip    = (const float*)d_in[3];
    const float* A_logs[2] = {(const float*)d_in[4], (const float*)d_in[5]};
    const float* conv_w  = (const float*)d_in[6];
    const float* ca_fc1  = (const float*)d_in[7];
    const float* ca_fc2  = (const float*)d_in[8];
    const float* m_in_proj[2]  = {(const float*)d_in[9],  (const float*)d_in[17]};
    const float* m_conv_w[2]   = {(const float*)d_in[10], (const float*)d_in[18]};
    const float* m_conv_b[2]   = {(const float*)d_in[11], (const float*)d_in[19]};
    const float* m_x_proj[2]   = {(const float*)d_in[12], (const float*)d_in[20]};
    const float* m_dt_w[2]     = {(const float*)d_in[13], (const float*)d_in[21]};
    const float* m_dt_b[2]     = {(const float*)d_in[14], (const float*)d_in[22]};
    const float* m_D[2]        = {(const float*)d_in[15], (const float*)d_in[23]};
    const float* m_out_proj[2] = {(const float*)d_in[16], (const float*)d_in[24]};

    float* ws = (float*)d_ws;
    size_t off = 0;
    auto alloc = [&](size_t n) { float* p = ws + off; off += (n + 63) & ~(size_t)63; return p; };

    float* xn      = alloc((size_t)B * L * C);
    float* P2      = alloc((size_t)B * 2048 * C);
    float* P4      = alloc((size_t)B * 1024 * C);
    float* xh      = alloc((size_t)B * 8192 * DM);
    float* attnM   = alloc((size_t)B * 4 * DM);
    float* oattn   = alloc((size_t)B * 4 * DM);
    float* xin     = alloc((size_t)B * L * DM);
    float* xz      = alloc((size_t)B * L * 2 * DI);
    float* xc      = alloc((size_t)B * L * DI);
    float* xdbl    = alloc((size_t)B * L * 40);
    float* dtb     = alloc((size_t)B * L * DI);
    float* ysc     = alloc((size_t)B * L * DI);
    float* gbuf    = alloc((size_t)B * L * DI);
    float* ydir    = alloc((size_t)B * L * DM);
    float* yacc    = alloc((size_t)B * L * DM);
    float* opre    = alloc((size_t)B * L * C);

    auto blocks1d = [](size_t total) { return (unsigned)((total + 255) / 256); };
    auto gemm = [&](const float* A, const float* W, float* O, int M, int N, int K, int act) {
        dim3 grid((unsigned)(M / 32), (unsigned)((N + 127) / 128));
        gemm_wmma_kernel<<<grid, 256, 0, stream>>>(A, W, O, M, N, K, act);
    };

    // 1) Input LayerNorm
    ln_kernel<<<B * L, 256, 0, stream>>>(x, norm_w, norm_b, xn, C);

    // 2) Pooled inputs (pool-before-GEMM fusion; df=4 shared by scales 2,3)
    pool_kernel<<<blocks1d((size_t)B * 2048 * C), 256, 0, stream>>>(xn, P2, B, L, C, 2);
    pool_kernel<<<blocks1d((size_t)B * 1024 * C), 256, 0, stream>>>(xn, P4, B, L, C, 4);

    // 3) Per-scale 1x1 conv (WMMA GEMM) + ReLU, then channel attention
    const float* Pp[4] = {xn, P2, P4, P4};
    float* xib[4];
    {
        size_t base = 0;
        for (int i = 0; i < 4; ++i) { xib[i] = xh + base; base += (size_t)B * Li[i] * DM; }
    }
    for (int i = 0; i < 4; ++i) {
        int M = B * Li[i];
        gemm(Pp[i], conv_w + (size_t)i * DM * C, xib[i], M, DM, C, /*relu*/1);
        chattn_kernel<<<B, 128, 0, stream>>>(xib[i], ca_fc1 + (size_t)i * DM * DM,
                                             ca_fc2 + (size_t)i * DM * DM, attnM, Li[i], i);
    }
    sattn_kernel<<<B, 128, 0, stream>>>(attnM, oattn);

    // 4) Bidirectional Mamba per scale
    for (int i = 0; i < 4; ++i) {
        int Lx = Li[i];
        int M  = B * Lx;
        for (int dir = 0; dir < 2; ++dir) {
            flipcopy_kernel<<<blocks1d((size_t)M * DM), 256, 0, stream>>>(
                xib[i], xin, B, Lx, DM, dir);
            // in_proj: (M,128) x (512,128)^T
            gemm(xin, m_in_proj[dir], xz, M, 2 * DI, DM, 0);
            // depthwise causal conv + SiLU
            conv_silu_kernel<<<blocks1d((size_t)M * DI), 256, 0, stream>>>(
                xz, m_conv_w[dir], m_conv_b[dir], xc, B, Lx);
            // x_proj: (M,256) x (40,256)^T
            gemm(xc, m_x_proj[dir], xdbl, M, 40, DI, 0);
            // dt = softplus(dt_in @ dt_w^T + dt_b)
            dt_kernel<<<blocks1d((size_t)M * DI), 256, 0, stream>>>(
                xdbl, m_dt_w[dir], m_dt_b[dir], dtb, (size_t)M);
            // selective scan
            scan_kernel<<<B, 256, 0, stream>>>(
                dtb, xc, xdbl, A_logs[dir] + (size_t)i * DI * 16, m_D[dir], ysc, Lx);
            // gating
            gate_kernel<<<blocks1d((size_t)M * DI), 256, 0, stream>>>(ysc, xz, gbuf, (size_t)M);
            // out_proj: (M,256) x (128,256)^T
            gemm(gbuf, m_out_proj[dir], ydir, M, DM, DI, 0);
            // accumulate (dir 1 un-flips)
            accum_kernel<<<blocks1d((size_t)M * DM), 256, 0, stream>>>(ydir, yacc, B, Lx, dir);
        }
        // attention-weighted combine + skip + nearest upsample into channel slot i
        combine_kernel<<<blocks1d((size_t)B * L * DM), 256, 0, stream>>>(
            yacc, xib[i], oattn, skip, opre, B, L, Lx, df[i], i);
    }

    // 5) Output LayerNorm
    ln_kernel<<<B * L, 256, 0, stream>>>(opre, norm_w, norm_b, (float*)d_out, C);
}
